// TaxoClassGAT_16346645528743
// MI455X (gfx1250) — compile-verified, matrix operand order checked by reading, not run on hardware
//
#include <hip/hip_runtime.h>

// ---------------------------------------------------------------------------
// TaxoClass GAT pipeline for MI455X (gfx1250, wave32, WMMA).
// All matmuls use v_wmma_f32_16x16x32_bf16 (fp32 accumulate).
// All B operands stored transposed (N x K) so both A and B fragments load as
// contiguous 128-bit chunks. GEMM waves compute 32x64 tiles: two A fragments
// share four B fragments -> 8 WMMAs per 12 b128 loads per K-step.
// ---------------------------------------------------------------------------

typedef __attribute__((ext_vector_type(16))) __bf16 v16bf;
typedef __attribute__((ext_vector_type(8)))  float  v8f;

union FragBf { v16bf v; unsigned int u[8]; };

#define C_LABELS 4096
#define HIDDEN   512
#define NHEADS   4
#define HEADD    128

__device__ __forceinline__ unsigned short f2bf(float f) {
    unsigned int u = __builtin_bit_cast(unsigned int, f);
    unsigned int r = (u + 0x7FFFu + ((u >> 16) & 1u)) >> 16;   // RNE
    return (unsigned short)r;
}

__device__ __forceinline__ int kpos(int p, int half) {
    // 16-bit A/B fragment K packing: lanes 0-15: K={0..7,16..23}, lanes 16-31: +8
    return (p < 4 ? 2 * p : 8 + 2 * p) + 8 * half;
}

// ---------------------------------------------------------------------------
// fp32 -> bf16 elementwise convert
// ---------------------------------------------------------------------------
__global__ void cvt_bf16_kernel(const float* __restrict__ src,
                                unsigned short* __restrict__ dst, long n) {
    long i = (long)blockIdx.x * blockDim.x + threadIdx.x;
    if (i < n) dst[i] = f2bf(src[i]);
}

// fp32 (K x N) -> bf16 transposed (N x K)
__global__ void cvt_t_bf16_kernel(const float* __restrict__ src,
                                  unsigned short* __restrict__ dst,
                                  int K, int N) {
    long i = (long)blockIdx.x * blockDim.x + threadIdx.x;
    long total = (long)K * N;
    if (i < total) {
        int k = (int)(i / N), n = (int)(i % N);
        dst[(long)n * K + k] = f2bf(src[i]);
    }
}

// ---------------------------------------------------------------------------
// bf16 WMMA GEMM: C = act(A(MxK) @ B(KxN) + bias), B passed TRANSPOSED (NxK).
// Each wave computes a 32x64 tile (2 A fragments x 4 B fragments = 8 WMMA
// per K-step); 4 waves/block -> 32x256 per block. Optional f32 output and
// bf16 output (optionally transposed N x M for downstream fragment loads).
// act: 0 = none, 1 = relu
// ---------------------------------------------------------------------------
__global__ __launch_bounds__(128)
void gemm_bf16_wmma(const unsigned short* __restrict__ A,
                    const unsigned short* __restrict__ Bt,
                    float* __restrict__ Cf,
                    unsigned short* __restrict__ Cbf,
                    const float* __restrict__ bias,
                    int M, int N, int K, int act, int transposeCbf) {
    const int lane = threadIdx.x & 31;
    const int wave = threadIdx.x >> 5;
    const int m0 = blockIdx.y * 32;
    const int n0 = (blockIdx.x * 4 + wave) * 64;
    const int mrow = lane & 15;
    const int half = lane >> 4;

    v8f zero = {0.f, 0.f, 0.f, 0.f, 0.f, 0.f, 0.f, 0.f};
    v8f acc[2][4];
#pragma unroll
    for (int mi = 0; mi < 2; ++mi)
#pragma unroll
        for (int t = 0; t < 4; ++t) acc[mi][t] = zero;

    const unsigned short* Arow0 = A + (long)(m0 + mrow) * K;
    const unsigned short* Arow1 = A + (long)(m0 + 16 + mrow) * K;
    const unsigned short* Brow0 = Bt + (long)(n0 + 0 * 16 + mrow) * K;
    const unsigned short* Brow1 = Bt + (long)(n0 + 1 * 16 + mrow) * K;
    const unsigned short* Brow2 = Bt + (long)(n0 + 2 * 16 + mrow) * K;
    const unsigned short* Brow3 = Bt + (long)(n0 + 3 * 16 + mrow) * K;

    for (int kc = 0; kc < K; kc += 32) {
        if (kc + 32 < K) {   // hint next K chunk into L0 (global_prefetch_b8)
            __builtin_prefetch(Arow0 + kc + 32, 0, 1);
            __builtin_prefetch(Brow0 + kc + 32, 0, 1);
        }
        FragBf a0, a1;
#pragma unroll
        for (int p = 0; p < 8; ++p) {
            int k = kc + kpos(p, half);
            a0.u[p] = *(const unsigned int*)(Arow0 + k);
            a1.u[p] = *(const unsigned int*)(Arow1 + k);
        }
        FragBf b0, b1, b2, b3;
#pragma unroll
        for (int p = 0; p < 8; ++p) {
            int k = kc + kpos(p, half);
            b0.u[p] = *(const unsigned int*)(Brow0 + k);
            b1.u[p] = *(const unsigned int*)(Brow1 + k);
            b2.u[p] = *(const unsigned int*)(Brow2 + k);
            b3.u[p] = *(const unsigned int*)(Brow3 + k);
        }
        acc[0][0] = __builtin_amdgcn_wmma_f32_16x16x32_bf16(false, a0.v, false, b0.v, (short)0, acc[0][0], false, false);
        acc[1][0] = __builtin_amdgcn_wmma_f32_16x16x32_bf16(false, a1.v, false, b0.v, (short)0, acc[1][0], false, false);
        acc[0][1] = __builtin_amdgcn_wmma_f32_16x16x32_bf16(false, a0.v, false, b1.v, (short)0, acc[0][1], false, false);
        acc[1][1] = __builtin_amdgcn_wmma_f32_16x16x32_bf16(false, a1.v, false, b1.v, (short)0, acc[1][1], false, false);
        acc[0][2] = __builtin_amdgcn_wmma_f32_16x16x32_bf16(false, a0.v, false, b2.v, (short)0, acc[0][2], false, false);
        acc[1][2] = __builtin_amdgcn_wmma_f32_16x16x32_bf16(false, a1.v, false, b2.v, (short)0, acc[1][2], false, false);
        acc[0][3] = __builtin_amdgcn_wmma_f32_16x16x32_bf16(false, a0.v, false, b3.v, (short)0, acc[0][3], false, false);
        acc[1][3] = __builtin_amdgcn_wmma_f32_16x16x32_bf16(false, a1.v, false, b3.v, (short)0, acc[1][3], false, false);
    }

#pragma unroll
    for (int mi = 0; mi < 2; ++mi) {
#pragma unroll
        for (int t = 0; t < 4; ++t) {
#pragma unroll
            for (int v = 0; v < 8; ++v) {
                int r = m0 + mi * 16 + v + 8 * half;
                int c = n0 + t * 16 + mrow;
                float x = acc[mi][t][v];
                if (bias) x += bias[c];
                if (act == 1) x = x > 0.f ? x : 0.f;
                if (Cf)  Cf[(long)r * N + c] = x;
                if (Cbf) {
                    unsigned short bv = f2bf(x);
                    if (transposeCbf) Cbf[(long)c * M + r] = bv;
                    else              Cbf[(long)r * N + c] = bv;
                }
            }
        }
    }
}

// ---------------------------------------------------------------------------
// Per-node attention scores: a_s[h][c] = <h[c, h*128: ], a_src[h]>, same a_d.
// ---------------------------------------------------------------------------
__global__ __launch_bounds__(128)
void attn_scores_kernel(const float* __restrict__ h,
                        const float* __restrict__ a_src,
                        const float* __restrict__ a_dst,
                        float* __restrict__ as_out,
                        float* __restrict__ ad_out) {
    __shared__ float red[128];
    const int c = blockIdx.x;
    const int t = threadIdx.x;
    for (int hh = 0; hh < NHEADS; ++hh) {
        float x = h[(long)c * HIDDEN + hh * HEADD + t];
        float vs = x * a_src[hh * HEADD + t];
        float vd = x * a_dst[hh * HEADD + t];

        red[t] = vs; __syncthreads();
        for (int s = 64; s > 0; s >>= 1) { if (t < s) red[t] += red[t + s]; __syncthreads(); }
        if (t == 0) as_out[hh * C_LABELS + c] = red[0];
        __syncthreads();

        red[t] = vd; __syncthreads();
        for (int s = 64; s > 0; s >>= 1) { if (t < s) red[t] += red[t + s]; __syncthreads(); }
        if (t == 0) ad_out[hh * C_LABELS + c] = red[0];
        __syncthreads();
    }
}

// ---------------------------------------------------------------------------
// Softmax statistics (online max + denom) per (head,row) over masked scores.
// ---------------------------------------------------------------------------
__global__ __launch_bounds__(256)
void attn_stats_kernel(const float* __restrict__ as_in,
                       const float* __restrict__ ad_in,
                       const float* __restrict__ adj,
                       float* __restrict__ Mv,
                       float* __restrict__ Sv) {
    __shared__ float sm[256], ss[256];
    const int c = blockIdx.x;
    const int t = threadIdx.x;
    const float* adjRow = adj + (long)c * C_LABELS;
    for (int hh = 0; hh < NHEADS; ++hh) {
        float asv = as_in[hh * C_LABELS + c];
        float m = -INFINITY, s = 0.f;
        for (int j = t; j < C_LABELS; j += 256) {
            if (adjRow[j] != 0.f) {
                float e = asv + ad_in[hh * C_LABELS + j];
                e = e > 0.f ? e : 0.2f * e;                     // leaky_relu
                if (e > m) { s = s * __expf(m - e) + 1.f; m = e; }
                else       { s += __expf(e - m); }
            }
        }
        sm[t] = m; ss[t] = s; __syncthreads();
        for (int st = 128; st > 0; st >>= 1) {
            if (t < st) {
                float m1 = sm[t], s1 = ss[t];
                float m2 = sm[t + st], s2 = ss[t + st];
                float mm = m1 > m2 ? m1 : m2;
                float sn = (mm == -INFINITY) ? 0.f
                         : s1 * __expf(m1 - mm) + s2 * __expf(m2 - mm);
                sm[t] = mm; ss[t] = sn;
            }
            __syncthreads();
        }
        if (t == 0) { Mv[hh * C_LABELS + c] = sm[0]; Sv[hh * C_LABELS + c] = ss[0]; }
        __syncthreads();
    }
}

// ---------------------------------------------------------------------------
// Attention aggregation via WMMA: out[i, h*128+d] = sum_j P[h,i,j]*h[j, h*128+d]
// hbfT is the head-feature matrix TRANSPOSED (HIDDEN x C) in bf16, so the
// B fragment loads are contiguous b128. One wave per (16-row block, head);
// the 16x32 bf16 probability tile is built in registers (exp/normalize/mask),
// then 8 WMMAs cover D=128.  act: 0 = none, 1 = elu.
// ---------------------------------------------------------------------------
__global__ __launch_bounds__(32)
void attn_aggregate_wmma(const float* __restrict__ as_in,
                         const float* __restrict__ ad_in,
                         const float* __restrict__ Mv,
                         const float* __restrict__ Sv,
                         const float* __restrict__ adj,
                         const unsigned short* __restrict__ hbfT,
                         float* __restrict__ outF,
                         unsigned short* __restrict__ outBf,
                         int act) {
    const int hh = blockIdx.y;
    const int i0 = blockIdx.x * 16;
    const int lane = threadIdx.x;
    const int mrow = lane & 15;
    const int half = lane >> 4;
    const int row = i0 + mrow;

    const float asv  = as_in[hh * C_LABELS + row];
    const float Mr   = Mv[hh * C_LABELS + row];
    const float Sr   = Sv[hh * C_LABELS + row];
    const float Sinv = (Sr > 0.f) ? 1.f / Sr : 0.f;
    const float* adjRow = adj + (long)row * C_LABELS;

    v8f zero = {0.f, 0.f, 0.f, 0.f, 0.f, 0.f, 0.f, 0.f};
    v8f acc[8];
#pragma unroll
    for (int nt = 0; nt < 8; ++nt) acc[nt] = zero;

    for (int kc = 0; kc < C_LABELS; kc += 32) {
        FragBf a;
#pragma unroll
        for (int p = 0; p < 8; ++p) {
            int k = kpos(p, half);
            int j = kc + k;
            float e0 = asv + ad_in[hh * C_LABELS + j];
            e0 = e0 > 0.f ? e0 : 0.2f * e0;
            float p0 = (adjRow[j] != 0.f) ? __expf(e0 - Mr) * Sinv : 0.f;
            float e1 = asv + ad_in[hh * C_LABELS + j + 1];
            e1 = e1 > 0.f ? e1 : 0.2f * e1;
            float p1 = (adjRow[j + 1] != 0.f) ? __expf(e1 - Mr) * Sinv : 0.f;
            a.u[p] = (unsigned int)f2bf(p0) | ((unsigned int)f2bf(p1) << 16);
        }
#pragma unroll
        for (int nt = 0; nt < 8; ++nt) {
            FragBf b;
            const unsigned short* Brow =
                hbfT + (long)(hh * HEADD + nt * 16 + mrow) * C_LABELS + kc;
#pragma unroll
            for (int p = 0; p < 8; ++p)
                b.u[p] = *(const unsigned int*)(Brow + kpos(p, half));
            acc[nt] = __builtin_amdgcn_wmma_f32_16x16x32_bf16(
                false, a.v, false, b.v, (short)0, acc[nt], false, false);
        }
    }

#pragma unroll
    for (int nt = 0; nt < 8; ++nt) {
#pragma unroll
        for (int v = 0; v < 8; ++v) {
            int r = i0 + v + 8 * half;
            int col = hh * HEADD + nt * 16 + mrow;
            float x = acc[nt][v];
            if (act == 1) x = x > 0.f ? x : __expf(x) - 1.f;     // elu
            if (outF) outF[(long)r * HIDDEN + col] = x;
            if (outBf) outBf[(long)r * HIDDEN + col] = f2bf(x);
        }
    }
}

// ---------------------------------------------------------------------------
// Host-side orchestration
// ---------------------------------------------------------------------------
extern "C" void kernel_launch(void* const* d_in, const int* in_sizes, int n_in,
                              void* d_out, int out_size, void* d_ws, size_t ws_size,
                              hipStream_t stream) {
    const float* doc_x      = (const float*)d_in[0];   // 16384 x 768
    const float* label_init = (const float*)d_in[1];   // 4096 x 768
    const float* W1         = (const float*)d_in[2];   // 768 x 512
    const float* a_src1     = (const float*)d_in[3];   // 4 x 128
    const float* a_dst1     = (const float*)d_in[4];
    const float* W2         = (const float*)d_in[5];   // 512 x 512
    const float* a_src2     = (const float*)d_in[6];
    const float* a_dst2     = (const float*)d_in[7];
    const float* doc_W      = (const float*)d_in[8];   // 768 x 512
    const float* doc_b      = (const float*)d_in[9];   // 512
    const float* adj        = (const float*)d_in[10];  // 4096 x 4096

    char* cur = (char*)d_ws;
    auto alloc = [&](size_t bytes) -> void* {
        void* r = (void*)cur;
        cur += (bytes + 255) & ~(size_t)255;
        return r;
    };

    const long N_DOCS = 16384, DOC_DIM = 768, LBL_DIM = 768;
    unsigned short* lbl_bf  = (unsigned short*)alloc((size_t)C_LABELS * LBL_DIM * 2);
    unsigned short* w1T_bf  = (unsigned short*)alloc((size_t)HIDDEN * LBL_DIM * 2);   // 512 x 768
    unsigned short* w2T_bf  = (unsigned short*)alloc((size_t)HIDDEN * HIDDEN * 2);    // 512 x 512
    unsigned short* dwT_bf  = (unsigned short*)alloc((size_t)HIDDEN * DOC_DIM * 2);   // 512 x 768
    unsigned short* dx_bf   = (unsigned short*)alloc((size_t)N_DOCS * DOC_DIM * 2);
    float*          h1f     = (float*)alloc((size_t)C_LABELS * HIDDEN * 4);
    unsigned short* h1T_bf  = (unsigned short*)alloc((size_t)HIDDEN * C_LABELS * 2);  // 512 x 4096
    float* as1 = (float*)alloc((size_t)NHEADS * C_LABELS * 4);
    float* ad1 = (float*)alloc((size_t)NHEADS * C_LABELS * 4);
    float* M1  = (float*)alloc((size_t)NHEADS * C_LABELS * 4);
    float* S1  = (float*)alloc((size_t)NHEADS * C_LABELS * 4);
    unsigned short* act1bf  = (unsigned short*)alloc((size_t)C_LABELS * HIDDEN * 2);  // 4096 x 512
    float*          h2f     = (float*)alloc((size_t)C_LABELS * HIDDEN * 4);
    unsigned short* h2T_bf  = (unsigned short*)alloc((size_t)HIDDEN * C_LABELS * 2);  // 512 x 4096
    float* as2 = (float*)alloc((size_t)NHEADS * C_LABELS * 4);
    float* ad2 = (float*)alloc((size_t)NHEADS * C_LABELS * 4);
    float* M2  = (float*)alloc((size_t)NHEADS * C_LABELS * 4);
    float* S2  = (float*)alloc((size_t)NHEADS * C_LABELS * 4);
    unsigned short* lblh_bf = (unsigned short*)alloc((size_t)C_LABELS * HIDDEN * 2);  // 4096 x 512 (= Bt for logits)
    unsigned short* doch_bf = (unsigned short*)alloc((size_t)N_DOCS * HIDDEN * 2);

    auto cvt = [&](const float* s, unsigned short* d, long n) {
        cvt_bf16_kernel<<<dim3((unsigned)((n + 255) / 256)), dim3(256), 0, stream>>>(s, d, n);
    };
    auto cvtT = [&](const float* s, unsigned short* d, int K, int N) {
        long n = (long)K * N;
        cvt_t_bf16_kernel<<<dim3((unsigned)((n + 255) / 256)), dim3(256), 0, stream>>>(s, d, K, N);
    };

    // bf16 conversions (weights transposed to N x K)
    cvt (label_init, lbl_bf, (long)C_LABELS * LBL_DIM);
    cvtT(W1,    w1T_bf, (int)LBL_DIM, HIDDEN);
    cvtT(W2,    w2T_bf, HIDDEN, HIDDEN);
    cvtT(doc_W, dwT_bf, (int)DOC_DIM, HIDDEN);
    cvt (doc_x, dx_bf, N_DOCS * DOC_DIM);

    // Layer 1: h1 = label_init @ W1   (bf16 copy written transposed for attention)
    gemm_bf16_wmma<<<dim3(HIDDEN / 256, C_LABELS / 32), dim3(128), 0, stream>>>(
        lbl_bf, w1T_bf, h1f, h1T_bf, nullptr, C_LABELS, HIDDEN, (int)LBL_DIM, 0, /*tC=*/1);
    attn_scores_kernel<<<dim3(C_LABELS), dim3(128), 0, stream>>>(h1f, a_src1, a_dst1, as1, ad1);
    attn_stats_kernel<<<dim3(C_LABELS), dim3(256), 0, stream>>>(as1, ad1, adj, M1, S1);
    // out1 = elu(attn @ h1), stored row-major bf16 (A of layer-2 GEMM)
    attn_aggregate_wmma<<<dim3(C_LABELS / 16, NHEADS), dim3(32), 0, stream>>>(
        as1, ad1, M1, S1, adj, h1T_bf, nullptr, act1bf, /*act=*/1);

    // Layer 2: h2 = elu(out1) @ W2   (bf16 copy transposed for attention)
    gemm_bf16_wmma<<<dim3(HIDDEN / 256, C_LABELS / 32), dim3(128), 0, stream>>>(
        act1bf, w2T_bf, h2f, h2T_bf, nullptr, C_LABELS, HIDDEN, HIDDEN, 0, /*tC=*/1);
    attn_scores_kernel<<<dim3(C_LABELS), dim3(128), 0, stream>>>(h2f, a_src2, a_dst2, as2, ad2);
    attn_stats_kernel<<<dim3(C_LABELS), dim3(256), 0, stream>>>(as2, ad2, adj, M2, S2);
    // label_h = attn @ h2, row-major bf16 (4096 x 512) == Bt of the logits GEMM
    attn_aggregate_wmma<<<dim3(C_LABELS / 16, NHEADS), dim3(32), 0, stream>>>(
        as2, ad2, M2, S2, adj, h2T_bf, nullptr, lblh_bf, /*act=*/0);

    // Doc path: doc_h = relu(doc_x @ doc_W + doc_b), bf16 row-major (A of logits)
    gemm_bf16_wmma<<<dim3(HIDDEN / 256, (unsigned)(N_DOCS / 32)), dim3(128), 0, stream>>>(
        dx_bf, dwT_bf, nullptr, doch_bf, doc_b, (int)N_DOCS, HIDDEN, (int)DOC_DIM, 1, 0);

    // logits = doc_h @ label_h^T  -> d_out (f32); Bt = label_h (4096 x 512)
    gemm_bf16_wmma<<<dim3(C_LABELS / 256, (unsigned)(N_DOCS / 32)), dim3(128), 0, stream>>>(
        doch_bf, lblh_bf, (float*)d_out, nullptr, nullptr,
        (int)N_DOCS, C_LABELS, HIDDEN, 0, 0);
}
